// PositionalAttention_60997125538238
// MI455X (gfx1250) — matrix operand order repaired and checked by reading, not compiled, not run on hardware
//
#include <hip/hip_runtime.h>

// Problem constants (match reference)
#define B_  32
#define S_  1024
#define D_  512
#define P_  20
#define G4  80      // 4*P
#define NT  16      // N-tiles (of 16) per wave in attention kernel (256 cols)

typedef __attribute__((ext_vector_type(16))) _Float16 v16h;
typedef __attribute__((ext_vector_type(8)))  float    v8f;

// ---------------------------------------------------------------------------
// Kernel A: tiled transpose + f32->f16 convert:  x[b][s][d] -> xT[b][d][s]
// So attention B-tiles (K along s, N along d) are contiguous v16h loads.
// ---------------------------------------------------------------------------
__global__ void k_transpose_xT(const float* __restrict__ x, _Float16* __restrict__ xT) {
    __shared__ _Float16 tile[64][68];   // padded to dodge bank conflicts
    const int b  = blockIdx.z;
    const int s0 = blockIdx.y * 64;
    const int d0 = blockIdx.x * 64;
    const int tid = threadIdx.x;        // 256 threads
    #pragma unroll
    for (int r = 0; r < 16; ++r) {
        int idx = tid + r * 256;
        int si = idx >> 6, dj = idx & 63;
        tile[si][dj] = (_Float16)x[((size_t)b * S_ + (s0 + si)) * D_ + (d0 + dj)];
    }
    __syncthreads();
    #pragma unroll
    for (int r = 0; r < 16; ++r) {
        int idx = tid + r * 256;
        int di = idx >> 6, sj = idx & 63;
        xT[((size_t)b * D_ + (d0 + di)) * S_ + (s0 + sj)] = tile[sj][di];
    }
}

// ---------------------------------------------------------------------------
// Kernel B: convert W_ih (80x512 f32) -> f16
// ---------------------------------------------------------------------------
__global__ void k_convert_w(const float* __restrict__ w, _Float16* __restrict__ wh, int n) {
    int i = blockIdx.x * blockDim.x + threadIdx.x;
    if (i < n) wh[i] = (_Float16)w[i];
}

// ---------------------------------------------------------------------------
// Kernel C: xw[m][g] = sum_d x[m][d] * W_ih[g][d]   (m = b*S+s flattened)
// One wave per 16x16 output tile; K=512 in 16 chunks of 32 via WMMA f16.
// A layout (16-bit 16x32): lane<16: M=lane, K=0..15 contiguous;
//                          lane>=16: M=lane-16, K=16..31 contiguous.
// B layout (16-bit 32x16): lane<16: N=lane, K=0..15 contiguous;
//                          lane>=16: N=lane-16, K=16..31 contiguous.
// ---------------------------------------------------------------------------
__global__ void k_gemm_xw(const float* __restrict__ x, const _Float16* __restrict__ wh,
                          float* __restrict__ xw) {
    const int lane  = threadIdx.x;        // 32
    const int m0    = blockIdx.y * 16;
    const int n0    = blockIdx.x * 16;
    const int hi    = lane >> 4;          // 0/1
    const int l15   = lane & 15;
    const int aoff  = hi ? 16 : 0;
    const int boff  = hi ? 16 : 0;
    const float*    ap = x  + (size_t)(m0 + l15) * D_;
    const _Float16* bp = wh + (size_t)(n0 + l15) * D_;

    v8f c = {};
    for (int kc = 0; kc < D_; kc += 32) {
        v16h a, bb;
        const float* a0 = ap + kc + aoff;
        #pragma unroll
        for (int i = 0; i < 8; ++i) {
            a[i]     = (_Float16)a0[i];
            a[8 + i] = (_Float16)a0[8 + i];
        }
        bb = *(const v16h*)(bp + kc + boff);
        c = __builtin_amdgcn_wmma_f32_16x16x32_f16(false, a, false, bb,
                                                   (short)0, c, false, false);
    }
    // C/D layout: VGPR r -> row m0 + r + (hi?8:0), col n0 + l15
    #pragma unroll
    for (int r = 0; r < 8; ++r) {
        int m = m0 + r + (hi ? 8 : 0);
        xw[(size_t)m * G4 + (n0 + l15)] = c[r];
    }
}

// ---------------------------------------------------------------------------
// Kernel D: sequential LSTM + mu/sigma recurrence. One block per batch.
// Outputs mus[b][s] and inv2s2[b][s] = 1/(2*sigma^2 + 0.001).
// ---------------------------------------------------------------------------
__global__ void k_lstm(const float* __restrict__ xw, const int* __restrict__ pad_lengths,
                       const float* __restrict__ W_hh, const float* __restrict__ b_ih,
                       const float* __restrict__ b_hh,
                       const float* __restrict__ W_mu, const float* __restrict__ b_mu,
                       const float* __restrict__ W_sig, const float* __restrict__ b_sig,
                       float* __restrict__ mus, float* __restrict__ inv2s2) {
    __shared__ float sWhh[G4 * P_];
    __shared__ float sbias[G4];
    __shared__ float sWmu[3 * P_];
    __shared__ float sWsig[P_];
    __shared__ float sh[P_];
    __shared__ float sgates[G4];

    const int b   = blockIdx.x;
    const int tid = threadIdx.x;   // 96
    for (int i = tid; i < G4 * P_; i += blockDim.x) sWhh[i] = W_hh[i];
    for (int i = tid; i < G4;      i += blockDim.x) sbias[i] = b_ih[i] + b_hh[i];
    for (int i = tid; i < 3 * P_;  i += blockDim.x) sWmu[i]  = W_mu[i];
    for (int i = tid; i < P_;      i += blockDim.x) sWsig[i] = W_sig[i];
    if (tid < P_) sh[tid] = 0.f;
    __syncthreads();

    float cstate  = 0.f;   // thread p (<20) owns c[p]
    float prev_mu = 0.f;   // thread 0
    const float invL = 1.f / (float)pad_lengths[b];
    const float* xwb = xw + (size_t)b * S_ * G4;

    for (int t = 0; t < S_; ++t) {
        if (tid < G4) {
            float acc = xwb[(size_t)t * G4 + tid] + sbias[tid];
            #pragma unroll
            for (int p = 0; p < P_; ++p) acc += sh[p] * sWhh[tid * P_ + p];
            sgates[tid] = acc;
        }
        __syncthreads();
        if (tid < P_) {
            float ig = 1.f / (1.f + __expf(-sgates[tid]));
            float fg = 1.f / (1.f + __expf(-sgates[P_ + tid]));
            float gg = tanhf(sgates[2 * P_ + tid]);
            float og = 1.f / (1.f + __expf(-sgates[3 * P_ + tid]));
            cstate = fg * cstate + ig * gg;
            sh[tid] = og * tanhf(cstate);
        }
        __syncthreads();
        if (tid == 0) {
            float m0a = b_mu[0], m1a = b_mu[1], m2a = b_mu[2], sg = b_sig[0];
            #pragma unroll
            for (int p = 0; p < P_; ++p) {
                float h = sh[p];
                m0a += h * sWmu[p];
                m1a += h * sWmu[P_ + p];
                m2a += h * sWmu[2 * P_ + p];
                sg  += h * sWsig[p];
            }
            m0a = fmaxf(m0a, 0.f); m1a = fmaxf(m1a, 0.f); m2a = fmaxf(m2a, 0.f);
            float sig = 1.f / (1.f + __expf(-sg));
            float mu  = m0a * prev_mu + m1a * invL + m2a * (float)(t + 1) * invL;
            mu = fmaxf(mu, (float)t * invL);
            prev_mu = mu;
            mus[(size_t)b * S_ + t]    = mu;
            inv2s2[(size_t)b * S_ + t] = 1.f / (2.f * sig * sig + 0.001f);
        }
        // thread 0 only READS sh here; next write to sh is after the next
        // barrier, so no extra sync needed.
    }
}

// ---------------------------------------------------------------------------
// Kernel E: fused Gaussian-attention  context = normalize(w) @ x.
// One wave per (b, 16-row j tile, NT*16 feature columns). Scores are
// generated in-register directly in the WMMA A layout; causality masked
// per element; L1 row-sums kept per lane and reduced with shfl_xor(16).
// ---------------------------------------------------------------------------
__global__ void k_attn(const _Float16* __restrict__ xT,
                       const float* __restrict__ mus, const float* __restrict__ inv2s2,
                       const int* __restrict__ pad_lengths,
                       float* __restrict__ out) {
    const int lane = threadIdx.x;          // 32
    const int b    = blockIdx.z;
    const int j0   = blockIdx.y * 16;
    const int n0   = blockIdx.x * (NT * 16);
    const int hi   = lane >> 4;
    const int l15  = lane & 15;
    const int row  = j0 + l15;             // this lane's score row

    const float invL = 1.f / (float)pad_lengths[b];
    const float mu   = mus[(size_t)b * S_ + row];
    // fold -1/(2*sigma^2+0.001) and log2(e) into one constant: w = exp2(d^2*c2)
    const float c2   = -inv2s2[(size_t)b * S_ + row] * 1.4426950408889634f;

    v8f zero = {};
    v8f c[NT];
    #pragma unroll
    for (int t = 0; t < NT; ++t) c[t] = zero;

    float rsum = 0.f;
    const _Float16* xb = xT + (size_t)b * D_ * S_;
    const int aoff = hi ? 16 : 0;
    const int kend = j0 + 16;              // causal: k <= j0+15

    for (int k0 = 0; k0 < kend; k0 += 32) {
        // --- generate A (Gaussian weights) in WMMA 16-bit A layout ---
        v16h a;
        #pragma unroll
        for (int i = 0; i < 8; ++i) {
            int k1 = k0 + aoff + i;
            float d1 = (float)k1 * invL - mu;
            float w1 = (k1 <= row) ? exp2f(d1 * d1 * c2) : 0.f;
            rsum += w1;
            a[i] = (_Float16)w1;
            int k2 = k1 + 8;
            float d2 = (float)k2 * invL - mu;
            float w2 = (k2 <= row) ? exp2f(d2 * d2 * c2) : 0.f;
            rsum += w2;
            a[8 + i] = (_Float16)w2;
        }
        // --- B tiles: contiguous 16 K-values per lane from xT ---
        const int kb = k0 + (hi ? 16 : 0);
        #pragma unroll
        for (int t = 0; t < NT; ++t) {
            const _Float16* bp = xb + (size_t)(n0 + t * 16 + l15) * S_ + kb;
            v16h bb = *(const v16h*)bp;
            __builtin_prefetch(bp + 32, 0, 1);   // global_prefetch_b8, next chunk
            c[t] = __builtin_amdgcn_wmma_f32_16x16x32_f16(false, a, false, bb,
                                                          (short)0, c[t], false, false);
        }
    }

    // L1 row-sum: lanes L and L+16 hold disjoint K halves of the same row.
    rsum += __shfl_xor(rsum, 16, 32);

    #pragma unroll
    for (int r = 0; r < 8; ++r) {
        int orow = r + (hi ? 8 : 0);                 // C layout row for VGPR r
        float rs = __shfl(rsum, orow, 32);           // full sum of that row
        float scale = 1.f / fmaxf(rs, 1e-12f);
        int j = j0 + orow;
        #pragma unroll
        for (int t = 0; t < NT; ++t) {
            out[((size_t)b * S_ + j) * D_ + (n0 + t * 16 + l15)] = c[t][r] * scale;
        }
    }
}

// ---------------------------------------------------------------------------
extern "C" void kernel_launch(void* const* d_in, const int* in_sizes, int n_in,
                              void* d_out, int out_size, void* d_ws, size_t ws_size,
                              hipStream_t stream) {
    (void)in_sizes; (void)n_in; (void)out_size; (void)ws_size;
    const float* x     = (const float*)d_in[0];
    const int*   pad   = (const int*)  d_in[1];
    const float* W_ih  = (const float*)d_in[2];
    const float* W_hh  = (const float*)d_in[3];
    const float* b_ih  = (const float*)d_in[4];
    const float* b_hh  = (const float*)d_in[5];
    const float* W_mu  = (const float*)d_in[6];
    const float* b_mu  = (const float*)d_in[7];
    const float* W_sig = (const float*)d_in[8];
    const float* b_sig = (const float*)d_in[9];
    float* out = (float*)d_out;

    // Workspace carve-up (~44.4 MB total)
    char* ws = (char*)d_ws;
    size_t off = 0;
    _Float16* xT = (_Float16*)(ws + off); off += (size_t)B_ * D_ * S_ * 2;   // 33.5 MB
    _Float16* wh = (_Float16*)(ws + off); off += (size_t)G4 * D_ * 2;
    off = (off + 255) & ~(size_t)255;
    float* xw  = (float*)(ws + off); off += (size_t)B_ * S_ * G4 * 4;        // 10.5 MB
    float* mus = (float*)(ws + off); off += (size_t)B_ * S_ * 4;
    float* i2s = (float*)(ws + off); off += (size_t)B_ * S_ * 4;

    k_transpose_xT<<<dim3(D_ / 64, S_ / 64, B_), 256, 0, stream>>>(x, xT);
    k_convert_w<<<(G4 * D_ + 255) / 256, 256, 0, stream>>>(W_ih, wh, G4 * D_);
    k_gemm_xw<<<dim3(G4 / 16, (B_ * S_) / 16), 32, 0, stream>>>(x, wh, xw);
    k_lstm<<<B_, 96, 0, stream>>>(xw, pad, W_hh, b_ih, b_hh,
                                  W_mu, b_mu, W_sig, b_sig, mus, i2s);
    k_attn<<<dim3(D_ / (16 * NT), S_ / 16, B_), 32, 0, stream>>>(xT, mus, i2s, pad, out);
}
